// CAM_15633680957452
// MI455X (gfx1250) — matrix operand order repaired and checked by reading, not compiled, not run on hardware
//
#include <hip/hip_runtime.h>
#include <stdint.h>

// CAM: out = gamma * (A @ softmax(A^T A, axis=-1)) + x,  A = x.reshape(B, HW, C)
// B=8, HW=16384, C=256.  fp32 in/out; matmuls on bf16 WMMA with f32 accum.
//
// Pipeline:
//   K0: xT[b][c][p] = bf16(x[b][p][c])        (LDS tile transpose; 67 MB, L2-resident)
//   K1: partial aTa from xT; all operand fetch = contiguous per-lane b128 loads
//   K2: reduce split-K partials + row softmax; store bf16 attnT[b][j][i]
//   K3: out = gamma * (A @ attn) + x          (A: fp32 float4 + cvt_pk; B: b128 bf16)

#define B_     8
#define HW_    16384
#define C_     256
#define SPLITK 16
#define PCHUNK (HW_ / SPLITK)   // 1024

typedef __attribute__((ext_vector_type(16))) __bf16 v16bf;
typedef __attribute__((ext_vector_type(8)))  __bf16 v8bf;
typedef __attribute__((ext_vector_type(4)))  __bf16 v4bf;
typedef __attribute__((ext_vector_type(8)))  float  v8f;

static __device__ __forceinline__ v16bf cat8(v8bf lo, v8bf hi) {
    return __builtin_shufflevector(lo, hi, 0, 1, 2, 3, 4, 5, 6, 7,
                                           8, 9, 10, 11, 12, 13, 14, 15);
}

static __device__ __forceinline__ void pack4(v16bf& v, int base, float4 f) {
    v[base + 0] = (__bf16)f.x;
    v[base + 1] = (__bf16)f.y;
    v[base + 2] = (__bf16)f.z;
    v[base + 3] = (__bf16)f.w;
}

// ---------------------------------------------------------------------------
// K0: fp32 [b][p][c] -> bf16 [b][c][p] via 64x64 LDS tile transpose.
// ---------------------------------------------------------------------------
__global__ __launch_bounds__(256) void k_transpose_bf16(const float* __restrict__ x,
                                                        __bf16* __restrict__ xT) {
    __shared__ __bf16 tile[64][68];     // +4 pad: dodge bank conflicts on transpose

    int blk = blockIdx.x;
    const int cb = blk & 3;   blk >>= 2;    // 4 c-blocks of 64
    const int pb = blk & 255; blk >>= 8;    // 256 p-blocks of 64
    const int b  = blk;

    const int p0 = pb * 64, c0 = cb * 64;
    const float* __restrict__ xb = x + (size_t)b * HW_ * C_;

    const int t  = threadIdx.x;
    const int pr = t >> 4;          // 0..15
    const int cq = (t & 15) * 4;    // 0..60
    #pragma unroll
    for (int it = 0; it < 4; ++it) {
        const int p = pr + it * 16;
        const float4 f = *(const float4*)(xb + (size_t)(p0 + p) * C_ + c0 + cq);
        tile[p][cq + 0] = (__bf16)f.x;
        tile[p][cq + 1] = (__bf16)f.y;
        tile[p][cq + 2] = (__bf16)f.z;
        tile[p][cq + 3] = (__bf16)f.w;
    }
    __syncthreads();

    __bf16* __restrict__ xTb = xT + (size_t)b * C_ * HW_;
    const int pq = (t & 15) * 4;    // contiguous p quad -> coalesced 8B stores
    const int cr = t >> 4;
    #pragma unroll
    for (int it = 0; it < 4; ++it) {
        const int c = cr + it * 16;
        v4bf o = { tile[pq + 0][c], tile[pq + 1][c], tile[pq + 2][c], tile[pq + 3][c] };
        *(v4bf*)(xTb + (size_t)(c0 + c) * HW_ + p0 + pq) = o;
    }
}

// ---------------------------------------------------------------------------
// K1: partial aTa.  Wave = one batch, 64(i) x 64(j) block, one split-K chunk.
// Operands from channel-major bf16 xT: every fetch is a contiguous per-lane
// 16B load in the same K order for A and B (A: K run [16h,+16);
// B: K run [16h,+16)); lanes lm / lm+16 share 64B lines.
// part layout: [s][b][i][j] (f32), written privately (deterministic).
// ---------------------------------------------------------------------------
__global__ __launch_bounds__(256) void k_ata_partial(const __bf16* __restrict__ xT,
                                                     float* __restrict__ part) {
    const int wave = blockIdx.x * (blockDim.x >> 5) + (threadIdx.x >> 5);
    const int lane = threadIdx.x & 31;
    const int half = lane >> 4;
    const int lm   = lane & 15;

    int t = wave;
    const int s  = t & 15; t >>= 4;   // split-K id
    const int jb = t & 3;  t >>= 2;   // j block (64 wide)
    const int ib = t & 3;  t >>= 2;   // i block (64 wide)
    const int b  = t;

    const int ibase = ib * 64;
    const int jbase = jb * 64;
    const __bf16* __restrict__ xb = xT + (size_t)b * C_ * HW_;

    v8f acc[4][4];
    {
        v8f z = {};
        #pragma unroll
        for (int i = 0; i < 4; ++i)
            #pragma unroll
            for (int j = 0; j < 4; ++j) acc[i][j] = z;
    }

    const int p_end = (s + 1) * PCHUNK;
    for (int p0 = s * PCHUNK; p0 < p_end; p0 += 32) {
        v16bf a[4], bm[4];
        #pragma unroll
        for (int ti = 0; ti < 4; ++ti) {
            const v8bf* rp = (const v8bf*)(xb + (size_t)(ibase + ti * 16 + lm) * HW_ + p0);
            a[ti] = cat8(rp[2 * half], rp[2 * half + 1]);   // K: [16h, +16) in order
        }
        #pragma unroll
        for (int tj = 0; tj < 4; ++tj) {
            const v8bf* rp = (const v8bf*)(xb + (size_t)(jbase + tj * 16 + lm) * HW_ +
                                           p0 + 16 * half);
            bm[tj] = cat8(rp[0], rp[1]);                    // K: [16h, +16) in order
        }
        #pragma unroll
        for (int ti = 0; ti < 4; ++ti)
            #pragma unroll
            for (int tj = 0; tj < 4; ++tj)
                acc[ti][tj] = __builtin_amdgcn_wmma_f32_16x16x32_bf16(
                    false, a[ti], false, bm[tj], (short)0, acc[ti][tj], false, false);
    }

    float* __restrict__ pp = part + ((size_t)s * B_ + b) * C_ * C_;
    #pragma unroll
    for (int ti = 0; ti < 4; ++ti)
        #pragma unroll
        for (int tj = 0; tj < 4; ++tj)
            #pragma unroll
            for (int r = 0; r < 8; ++r) {
                const int i = ibase + ti * 16 + half * 8 + r;   // C/D: M = r + 8*half
                const int j = jbase + tj * 16 + lm;             //      N = lm
                pp[(size_t)i * C_ + j] = acc[ti][tj][r];
            }
}

// ---------------------------------------------------------------------------
// K2: reduce split-K partials + row softmax; store TRANSPOSED bf16:
// attnT[b][j][i] = softmax_j(aTa[b][i][:])[j].  One wave per (b, i) row.
// ---------------------------------------------------------------------------
__global__ __launch_bounds__(256) void k_softmax_t(const float* __restrict__ part,
                                                   __bf16* __restrict__ attnT) {
    const int wave = blockIdx.x * (blockDim.x >> 5) + (threadIdx.x >> 5);
    const int lane = threadIdx.x & 31;
    const int b = wave >> 8;
    const int i = wave & 255;

    float v[8];
    #pragma unroll
    for (int t = 0; t < 8; ++t) {
        const int j = lane + 32 * t;
        float acc = 0.f;
        #pragma unroll
        for (int s = 0; s < SPLITK; ++s)
            acc += part[(((size_t)s * B_ + b) * C_ + i) * C_ + j];
        v[t] = acc;
    }
    float mx = v[0];
    #pragma unroll
    for (int t = 1; t < 8; ++t) mx = fmaxf(mx, v[t]);
    #pragma unroll
    for (int off = 16; off > 0; off >>= 1)
        mx = fmaxf(mx, __shfl_xor(mx, off, 32));
    float sm = 0.f;
    #pragma unroll
    for (int t = 0; t < 8; ++t) { v[t] = __expf(v[t] - mx); sm += v[t]; }
    #pragma unroll
    for (int off = 16; off > 0; off >>= 1)
        sm += __shfl_xor(sm, off, 32);
    const float inv = 1.0f / sm;
    #pragma unroll
    for (int t = 0; t < 8; ++t) {
        const int j = lane + 32 * t;
        attnT[((size_t)b * C_ + j) * C_ + i] = (__bf16)(v[t] * inv);
    }
}

// ---------------------------------------------------------------------------
// K3: out[b,p,j] = gamma * sum_c x[b,p,c] * attn[b,c,j] + x[b,p,j]
// Wave = 32(p) x 64(j) block, K over C=256 in steps of 32.
// A: fp32 float4 loads + cvt_pk (x also feeds the residual).
// B: contiguous per-lane b128 bf16 loads from attnT (element order = K order).
// ---------------------------------------------------------------------------
__global__ __launch_bounds__(256) void k_gemm2(const float* __restrict__ x,
                                               const __bf16* __restrict__ attnT,
                                               const float* __restrict__ gamma,
                                               float* __restrict__ out) {
    const int wave = blockIdx.x * (blockDim.x >> 5) + (threadIdx.x >> 5);
    const int lane = threadIdx.x & 31;
    const int half = lane >> 4;
    const int lm   = lane & 15;

    int t = wave;
    const int nb = t & 3;    t >>= 2;   // 64-wide j block
    const int pb = t & 511;  t >>= 9;   // 32-tall p block
    const int b  = t;

    const int pbase = pb * 32;
    const int nbase = nb * 64;
    const float g = gamma[0];

    const float* __restrict__ xb = x     + (size_t)b * HW_ * C_;
    const __bf16* __restrict__ at = attnT + (size_t)b * C_ * C_;

    v8f acc[2][4];
    {
        v8f z = {};
        #pragma unroll
        for (int i = 0; i < 2; ++i)
            #pragma unroll
            for (int j = 0; j < 4; ++j) acc[i][j] = z;
    }

    for (int k0 = 0; k0 < C_; k0 += 32) {
        v16bf a[2], bm[4];
        #pragma unroll
        for (int ti = 0; ti < 2; ++ti) {
            const float4* rp =
                (const float4*)(xb + (size_t)(pbase + ti * 16 + lm) * C_ + k0);
            pack4(a[ti], 0,  rp[half * 2 + 0]);     // K =  8h + 0..3
            pack4(a[ti], 4,  rp[half * 2 + 1]);     // K =  8h + 4..7
            pack4(a[ti], 8,  rp[4 + half * 2 + 0]); // K = 16 + 8h + 0..3
            pack4(a[ti], 12, rp[4 + half * 2 + 1]); // K = 16 + 8h + 4..7
        }
        #pragma unroll
        for (int tj = 0; tj < 4; ++tj) {
            const v8bf* rp =
                (const v8bf*)(at + (size_t)(nbase + tj * 16 + lm) * C_ + k0);
            bm[tj] = cat8(rp[half], rp[2 + half]);  // K = [8h,+8) U [16+8h,+8)
        }
        #pragma unroll
        for (int ti = 0; ti < 2; ++ti)
            #pragma unroll
            for (int tj = 0; tj < 4; ++tj)
                acc[ti][tj] = __builtin_amdgcn_wmma_f32_16x16x32_bf16(
                    false, a[ti], false, bm[tj], (short)0, acc[ti][tj], false, false);
    }

    float* __restrict__ ob = out + (size_t)b * HW_ * C_;
    #pragma unroll
    for (int ti = 0; ti < 2; ++ti)
        #pragma unroll
        for (int tj = 0; tj < 4; ++tj)
            #pragma unroll
            for (int r = 0; r < 8; ++r) {
                const int p = pbase + ti * 16 + half * 8 + r;
                const int j = nbase + tj * 16 + lm;
                const size_t idx = (size_t)p * C_ + j;
                ob[idx] = g * acc[ti][tj][r] + xb[idx];
            }
}

// ---------------------------------------------------------------------------
extern "C" void kernel_launch(void* const* d_in, const int* in_sizes, int n_in,
                              void* d_out, int out_size, void* d_ws, size_t ws_size,
                              hipStream_t stream) {
    (void)in_sizes; (void)n_in; (void)out_size; (void)ws_size;

    const float* x     = (const float*)d_in[0];
    const float* gamma = (const float*)d_in[1];
    float* out = (float*)d_out;

    // Workspace layout (bytes):
    //   xT    : B*C*HW bf16  = 67,108,864
    //   part  : SPLITK*B*C*C f32 = 33,554,432
    //   attnT : B*C*C bf16   =  1,048,576
    __bf16* xT    = (__bf16*)d_ws;
    float*  part  = (float*)((char*)d_ws + (size_t)B_ * C_ * HW_ * 2);
    __bf16* attnT = (__bf16*)((char*)part + (size_t)SPLITK * B_ * C_ * C_ * 4);

    // K0: 8(b) * 256(pb) * 4(cb) = 8192 blocks x 256 threads
    k_transpose_bf16<<<8192, 256, 0, stream>>>(x, xT);
    // K1: 8(b) * 4(ib) * 4(jb) * 16(s) = 2048 waves -> 256 blocks x 8 waves
    k_ata_partial<<<256, 256, 0, stream>>>(xT, part);
    // K2: 8(b) * 256(i) = 2048 waves -> 256 blocks x 8 waves
    k_softmax_t<<<256, 256, 0, stream>>>(part, attnT);
    // K3: 8(b) * 512(pb) * 4(nb) = 16384 waves -> 2048 blocks x 8 waves
    k_gemm2<<<2048, 256, 0, stream>>>(x, attnT, gamma, out);
}
